// DEQLayer_39453569581627
// MI455X (gfx1250) — matrix operand order repaired and checked by reading, not compiled, not run on hardware
//
#include <hip/hip_runtime.h>
#include <math.h>

typedef __bf16 bf16_t;
typedef __attribute__((ext_vector_type(16))) __bf16 v16bf;
typedef __attribute__((ext_vector_type(8)))  float  v8f;
typedef __attribute__((ext_vector_type(4)))  float  f4;

#define KMAX 11
#define NB 8
#define DHID 512
#define MROWS 8192
#define NTOT  4194304          // B*L*D
#define NPB   524288           // L*D
#define NTOT4 (NTOT/4)
#define NPB4  (NPB/4)
#define F_EPS 1e-5f

struct Ctrl {
  float low;
  float nrm2;
  float denom[NB];
  float coefA[KMAX*NB];
  float coefB[KMAX*NB];
  int cnt, done, improve, app, idx, step, cntA, cntB;
};

struct __align__(8) bf4 { bf16_t a, b, c, d; };

__device__ inline float waveSum(float v) {
#pragma unroll
  for (int o = 16; o > 0; o >>= 1) v += __shfl_xor(v, o, 32);
  return v;
}

// blockDim.x must be 256 (8 wave32s)
__device__ inline void blockAtomicSum(float v, float* dst, float* red) {
  int lane = threadIdx.x & 31, w = threadIdx.x >> 5;
  v = waveSum(v);
  if (lane == 0) red[w] = v;
  __syncthreads();
  if (threadIdx.x == 0) {
    float s = 0.f;
#pragma unroll
    for (int j = 0; j < 8; j++) s += red[j];
    atomicAdd(dst, s);
  }
  __syncthreads();
}

// ---- CDNA5 async global->LDS DMA (ASYNCcnt-tracked, no VGPR staging) ----
// LDS byte address = low 32 bits of the generic LDS-aperture address (ISA 10.2).
__device__ inline unsigned lds_off(const void* p) {
  return (unsigned)(unsigned long long)p;
}
__device__ inline void async_ld16(unsigned lds, unsigned long long gaddr) {
  asm volatile("global_load_async_to_lds_b128 %0, %1, off"
               :: "v"(lds), "v"(gaddr) : "memory");
}
__device__ inline void wait_async0() {
  asm volatile("s_wait_asynccnt 0x0" ::: "memory");
}

// ---------------- WMMA GEMM: out[8192,512] = A(bf16)[8192,512] @ Bt(bf16)[512,512]^T-stored ----
// Bt is stored "transposed": Bt[n*512 + k] = B[k][n], so tile loads are row-contiguous.
// mode 0: out = acc + bias[col]                       (X = E @ Winj^T + binj)
// mode 1: out = tanh(acc + bias[col]) + Xadd - xn,    plus ||out||^2 atomic into nrm2
__global__ __launch_bounds__(256)
void k_gemm(const bf16_t* __restrict__ A, const bf16_t* __restrict__ Bt,
            const float* __restrict__ bias, const float* __restrict__ Xadd,
            const float* __restrict__ xn, float* __restrict__ out,
            float* __restrict__ nrm2, int mode)
{
  __shared__ __align__(16) bf16_t Asl[2][64 * 40];    // double-buffered, pitch 40
  __shared__ __align__(16) bf16_t Bsl[2][128 * 40];
  __shared__ float red[8];

  const int nBase = blockIdx.x * 128;
  const int mBase = blockIdx.y * 64;
  const int tid   = threadIdx.x;
  const int w     = tid >> 5, lane = tid & 31;
  const int half  = lane >> 4, l16 = lane & 15;
  const int mW    = (w & 3) * 16;
  const int nW    = (w >> 2) * 64;

  v8f zero8 = 0.f;
  v8f acc[4] = { zero8, zero8, zero8, zero8 };

  // per-thread 16B chunk for tile staging
  const int rowA = tid >> 2;          // 0..63
  const int seg  = (tid & 3) * 8;     // bf16 elements: 0,8,16,24

  const unsigned long long gA  =
      (unsigned long long)(A  + (size_t)(mBase + rowA) * DHID + seg);
  const unsigned long long gB0 =
      (unsigned long long)(Bt + (size_t)(nBase + rowA) * DHID + seg);
  const unsigned long long gB1 =
      (unsigned long long)(Bt + (size_t)(nBase + rowA + 64) * DHID + seg);
  const unsigned lA[2]  = { lds_off(&Asl[0][rowA * 40 + seg]),
                            lds_off(&Asl[1][rowA * 40 + seg]) };
  const unsigned lB0[2] = { lds_off(&Bsl[0][rowA * 40 + seg]),
                            lds_off(&Bsl[1][rowA * 40 + seg]) };
  const unsigned lB1[2] = { lds_off(&Bsl[0][(rowA + 64) * 40 + seg]),
                            lds_off(&Bsl[1][(rowA + 64) * 40 + seg]) };

  // prologue: tile 0 -> buffer 0
  async_ld16(lA[0],  gA);
  async_ld16(lB0[0], gB0);
  async_ld16(lB1[0], gB1);

  for (int kb = 0; kb < DHID; kb += 32) {
    const int p = (kb >> 5) & 1;
    wait_async0();            // own async copies for buffer p complete
    __syncthreads();          // everyone's copies visible; prev compute done

    if (kb + 32 < DHID) {     // stream next tile into other buffer during compute
      const unsigned long long go = (unsigned long long)((kb + 32) * 2);
      async_ld16(lA[p ^ 1],  gA  + go);
      async_ld16(lB0[p ^ 1], gB0 + go);
      async_ld16(lB1[p ^ 1], gB1 + go);
    }

    // A fragment (16x32 bf16): lanes 0-15 -> K0=0, lanes 16-31 -> K0=8;
    // elements 0..7 = K0..K0+7, elements 8..15 = K0+16..K0+23
    union { v16bf v; f4 f[2]; } af;
    const bf16_t* ap = &Asl[p][(mW + l16) * 40 + half * 8];
    af.f[0] = *(const f4*)ap;
    af.f[1] = *(const f4*)(ap + 16);

#pragma unroll
    for (int t = 0; t < 4; t++) {
      // B fragment (32x16 bf16): lanes 0-15 -> K0=0, lanes 16-31 -> K0=8; elements 0..7 = K0..K0+7, 8..15 = K0+16..K0+23; N = l16
      union { v16bf v; f4 f[2]; } bfr;
      const bf16_t* bp = &Bsl[p][(nW + t * 16 + l16) * 40 + half * 8];
      bfr.f[0] = *(const f4*)bp;
      bfr.f[1] = *(const f4*)(bp + 16);
      acc[t] = __builtin_amdgcn_wmma_f32_16x16x32_bf16(
          false, af.v, false, bfr.v, (short)0, acc[t], false, false);
    }
  }

  float sq = 0.f;
#pragma unroll
  for (int t = 0; t < 4; t++) {
    const int col = nBase + nW + t * 16 + l16;
    const float bb = bias[col];
#pragma unroll
    for (int r = 0; r < 8; r++) {
      const int row = mBase + mW + half * 8 + r;   // C layout: lanes16-31 hold M=r+8
      const size_t o = (size_t)row * DHID + col;
      const float v = acc[t][r];
      if (mode == 0) {
        out[o] = v + bb;
      } else {
        const float g = tanhf(v + bb) + Xadd[o] - xn[o];
        out[o] = g;
        sq += g * g;
      }
    }
  }
  if (mode == 1) blockAtomicSum(sq, nrm2, red);
}

// ---------------- conversions ----------------
__global__ void k_cvt_tr(const float* __restrict__ in, bf16_t* __restrict__ out) {
  int i = blockIdx.x * 256 + threadIdx.x;          // i = n*512 + k
  int r = i >> 9, c = i & 511;
  out[i] = (bf16_t)in[c * DHID + r];               // out[n][k] = in[k][n]
}

__global__ void k_cvt_lin(const float* __restrict__ in, bf16_t* __restrict__ out, int n) {
  int i = blockIdx.x * 256 + threadIdx.x;
  if (i < n) out[i] = (bf16_t)in[i];
}

// ---------------- plane elementwise ----------------
__global__ __launch_bounds__(256)
void k_init(const float* __restrict__ zinit, float* __restrict__ x,
            float* __restrict__ lowx, bf16_t* __restrict__ zbf) {
  size_t i = (size_t)blockIdx.x * 256 + threadIdx.x;
  f4 v = ((const f4*)zinit)[i];
  ((f4*)x)[i] = v;
  ((f4*)lowx)[i] = v;
  bf4 p; p.a = (bf16_t)v.x; p.b = (bf16_t)v.y; p.c = (bf16_t)v.z; p.d = (bf16_t)v.w;
  ((bf4*)zbf)[i] = p;
}

__global__ __launch_bounds__(256)
void k_copy(const float* __restrict__ s, float* __restrict__ d) {
  size_t i = (size_t)blockIdx.x * 256 + threadIdx.x;
  ((f4*)d)[i] = ((const f4*)s)[i];
}

__global__ __launch_bounds__(256)
void k_xn(const float* __restrict__ x, const float* __restrict__ upd,
          float* __restrict__ xn, bf16_t* __restrict__ zbf) {
  size_t i = (size_t)blockIdx.x * 256 + threadIdx.x;
  f4 v = ((const f4*)x)[i] + ((const f4*)upd)[i];
  ((f4*)xn)[i] = v;
  bf4 p; p.a = (bf16_t)v.x; p.b = (bf16_t)v.y; p.c = (bf16_t)v.z; p.d = (bf16_t)v.w;
  ((bf4*)zbf)[i] = p;
}

// ---------------- control (1 thread) ----------------
__global__ void k_ctrl_init(Ctrl* c) {
  c->low = 3.4e38f; c->nrm2 = 0.f;
  for (int j = 0; j < NB; j++) c->denom[j] = 0.f;
  for (int j = 0; j < KMAX * NB; j++) { c->coefA[j] = 0.f; c->coefB[j] = 0.f; }
  c->cnt = 0; c->done = 0; c->improve = 0; c->app = 0;
  c->idx = 0; c->step = 1; c->cntA = 0; c->cntB = 0;
}

__global__ void k_setlow(Ctrl* c) { c->low = sqrtf(c->nrm2); }

__global__ void k_ctrl_pre(Ctrl* c) {
  c->nrm2 = 0.f;
  for (int j = 0; j < NB; j++) c->denom[j] = 0.f;
  for (int j = 0; j < KMAX * NB; j++) { c->coefA[j] = 0.f; c->coefB[j] = 0.f; }
}

__global__ void k_ctrl1(Ctrl* c, int i) {
  float nrm = sqrtf(c->nrm2);
  int done = c->done;
  int improve = (nrm < c->low) && !done;
  if (improve) c->low = nrm;
  c->improve = improve;
  int cnt = c->cnt;
  int app = (i >= 1) && (cnt < KMAX) && !done && (nrm >= F_EPS);
  c->app = app;
  c->idx = (cnt < KMAX - 1) ? cnt : (KMAX - 1);
  c->cntA = cnt;            // mask for Bdg / coefA
  c->cntB = cnt + app;      // mask for new_upd
  c->cnt  = cnt + app;
  c->step = !done;
  c->done = done || (nrm < F_EPS);
}

// ---------------- batch streaming kernels: grid (128, NB), 256 thr, 4 x f4/thr ----------
__global__ __launch_bounds__(256)
void k_dg_coef(const float* __restrict__ gxn, const float* __restrict__ gx,
               const float* __restrict__ xn, const float* __restrict__ V,
               float* __restrict__ dg, float* __restrict__ lowx, Ctrl* __restrict__ ctrl)
{
  __shared__ float red[8];
  const int b = blockIdx.y;
  const int cnt = ctrl->cntA;
  const int improve = ctrl->improve;
  const size_t base4 = (size_t)b * NPB4;
  float acc[KMAX];
#pragma unroll
  for (int k = 0; k < KMAX; k++) acc[k] = 0.f;
#pragma unroll
  for (int it = 0; it < 4; ++it) {
    size_t off = base4 + (size_t)blockIdx.x * 1024 + it * 256 + threadIdx.x;
    f4 gn = ((const f4*)gxn)[off];
    f4 d  = gn - ((const f4*)gx)[off];
    ((f4*)dg)[off] = d;
    if (improve) ((f4*)lowx)[off] = ((const f4*)xn)[off];
#pragma unroll
    for (int k = 0; k < KMAX; k++)
      if (k < cnt) {
        f4 vv = ((const f4*)V)[(size_t)k * NTOT4 + off];
        acc[k] += vv.x * d.x + vv.y * d.y + vv.z * d.z + vv.w * d.w;
      }
  }
#pragma unroll
  for (int k = 0; k < KMAX; k++)
    if (k < cnt) blockAtomicSum(acc[k], &ctrl->coefA[k * NB + b], red);
}

__global__ __launch_bounds__(256)
void k_bdg(const float* __restrict__ dg, const float* __restrict__ U,
           float* __restrict__ Bdg, Ctrl* __restrict__ ctrl)
{
  __shared__ float red[8];
  const int b = blockIdx.y;
  const int cnt = ctrl->cntA;
  float cf[KMAX];
#pragma unroll
  for (int k = 0; k < KMAX; k++) cf[k] = (k < cnt) ? ctrl->coefA[k * NB + b] : 0.f;
  const size_t base4 = (size_t)b * NPB4;
  float den = 0.f;
#pragma unroll
  for (int it = 0; it < 4; ++it) {
    size_t off = base4 + (size_t)blockIdx.x * 1024 + it * 256 + threadIdx.x;
    f4 d = ((const f4*)dg)[off];
    f4 s = -d;
#pragma unroll
    for (int k = 0; k < KMAX; k++)
      if (k < cnt) s += ((const f4*)U)[(size_t)k * NTOT4 + off] * cf[k];
    ((f4*)Bdg)[off] = s;
    den += d.x * s.x + d.y * s.y + d.z * s.z + d.w * s.w;
  }
  blockAtomicSum(den, &ctrl->denom[b], red);
}

__global__ __launch_bounds__(256)
void k_uv(const float* __restrict__ xn, const float* __restrict__ x,
          const float* __restrict__ Bdg, const float* __restrict__ dg,
          float* __restrict__ U, float* __restrict__ V, Ctrl* __restrict__ ctrl)
{
  if (!ctrl->app) return;
  const int b = blockIdx.y;
  const float inv = 1.f / ctrl->denom[b];
  const size_t pl = (size_t)ctrl->idx * NTOT4;
  const size_t base4 = (size_t)b * NPB4;
#pragma unroll
  for (int it = 0; it < 4; ++it) {
    size_t off = base4 + (size_t)blockIdx.x * 1024 + it * 256 + threadIdx.x;
    f4 u = (((const f4*)xn)[off] - ((const f4*)x)[off] - ((const f4*)Bdg)[off]) * inv;
    ((f4*)U)[pl + off] = u;
    ((f4*)V)[pl + off] = ((const f4*)dg)[off];
  }
}

__global__ __launch_bounds__(256)
void k_coef2(const float* __restrict__ gxn, const float* __restrict__ V,
             Ctrl* __restrict__ ctrl)
{
  __shared__ float red[8];
  const int b = blockIdx.y;
  const int cnt = ctrl->cntB;
  const size_t base4 = (size_t)b * NPB4;
  float acc[KMAX];
#pragma unroll
  for (int k = 0; k < KMAX; k++) acc[k] = 0.f;
#pragma unroll
  for (int it = 0; it < 4; ++it) {
    size_t off = base4 + (size_t)blockIdx.x * 1024 + it * 256 + threadIdx.x;
    f4 g = ((const f4*)gxn)[off];
#pragma unroll
    for (int k = 0; k < KMAX; k++)
      if (k < cnt) {
        f4 vv = ((const f4*)V)[(size_t)k * NTOT4 + off];
        acc[k] += vv.x * g.x + vv.y * g.y + vv.z * g.z + vv.w * g.w;
      }
  }
#pragma unroll
  for (int k = 0; k < KMAX; k++)
    if (k < cnt) blockAtomicSum(acc[k], &ctrl->coefB[k * NB + b], red);
}

__global__ __launch_bounds__(256)
void k_state(const float* __restrict__ xn, const float* __restrict__ gxn,
             const float* __restrict__ U, float* __restrict__ x,
             float* __restrict__ gx, float* __restrict__ upd,
             Ctrl* __restrict__ ctrl, float sgn)
{
  const int b = blockIdx.y;
  const int step = ctrl->step;
  const int cnt = ctrl->cntB;
  float cf[KMAX];
#pragma unroll
  for (int k = 0; k < KMAX; k++) cf[k] = (k < cnt) ? ctrl->coefB[k * NB + b] : 0.f;
  const size_t base4 = (size_t)b * NPB4;
#pragma unroll
  for (int it = 0; it < 4; ++it) {
    size_t off = base4 + (size_t)blockIdx.x * 1024 + it * 256 + threadIdx.x;
    f4 gn = ((const f4*)gxn)[off];
    f4 nu = gn * sgn;
#pragma unroll
    for (int k = 0; k < KMAX; k++)
      if (k < cnt) nu += ((const f4*)U)[(size_t)k * NTOT4 + off] * cf[k];
    if (step) {
      ((f4*)x)[off]   = ((const f4*)xn)[off];
      ((f4*)gx)[off]  = gn;
      ((f4*)upd)[off] = nu;
    }
  }
}

// ---------------- host orchestration ----------------
extern "C" void kernel_launch(void* const* d_in, const int* in_sizes, int n_in,
                              void* d_out, int out_size, void* d_ws, size_t ws_size,
                              hipStream_t stream)
{
  (void)in_sizes; (void)n_in; (void)out_size; (void)ws_size;
  const float* E     = (const float*)d_in[0];
  const float* zinit = (const float*)d_in[1];
  const float* Wf    = (const float*)d_in[2];
  const float* bfv   = (const float*)d_in[3];
  const float* Winj  = (const float*)d_in[4];
  const float* binj  = (const float*)d_in[5];

  float* ws   = (float*)d_ws;
  float* X    = ws;
  float* x    = ws + (size_t)1  * NTOT;
  float* gx   = ws + (size_t)2  * NTOT;
  float* upd  = ws + (size_t)3  * NTOT;
  float* xn   = ws + (size_t)4  * NTOT;
  float* gxn  = ws + (size_t)5  * NTOT;
  float* dg   = ws + (size_t)6  * NTOT;
  float* Bdg  = ws + (size_t)7  * NTOT;
  float* lowx = ws + (size_t)8  * NTOT;
  float* U    = ws + (size_t)9  * NTOT;   // [KMAX][NTOT]
  float* V    = ws + (size_t)20 * NTOT;   // [KMAX][NTOT]
  bf16_t* zbf   = (bf16_t*)(ws + (size_t)31 * NTOT);
  bf16_t* WfT   = zbf + NTOT;
  bf16_t* WinjB = WfT + DHID * DHID;
  Ctrl*   ctrl  = (Ctrl*)(WinjB + DHID * DHID);

  dim3 tb(256);
  dim3 gGemm(4, 128);                 // N/128, M/64
  dim3 gPlane(NTOT4 / 256);           // 4096
  dim3 gBatch(NPB4 / 1024, NB);       // (128, 8)

  // constants: bf16 weights (Wf transposed for row-contiguous B tiles; Winj as-is == (Winj^T)^T)
  k_cvt_tr <<<DHID * DHID / 256, tb, 0, stream>>>(Wf, WfT);
  k_cvt_lin<<<DHID * DHID / 256, tb, 0, stream>>>(Winj, WinjB, DHID * DHID);
  // X = E @ Winj^T + binj  (use zbf as scratch for bf16(E))
  k_cvt_lin<<<NTOT / 256, tb, 0, stream>>>(E, zbf, NTOT);
  k_gemm<<<gGemm, tb, 0, stream>>>(zbf, WinjB, binj, nullptr, nullptr, X, nullptr, 0);

  // broyden init: x = lowx = z_init; gx0 = g(x0); low = ||gx0||; upd = gx0
  k_ctrl_init<<<1, 1, 0, stream>>>(ctrl);
  k_init<<<gPlane, tb, 0, stream>>>(zinit, x, lowx, zbf);
  k_gemm<<<gGemm, tb, 0, stream>>>(zbf, WfT, bfv, X, x, gx, &ctrl->nrm2, 1);
  k_setlow<<<1, 1, 0, stream>>>(ctrl);
  k_copy<<<gPlane, tb, 0, stream>>>(gx, upd);

  for (int i = 0; i < 12; i++) {
    k_ctrl_pre<<<1, 1, 0, stream>>>(ctrl);
    k_xn<<<gPlane, tb, 0, stream>>>(x, upd, xn, zbf);
    k_gemm<<<gGemm, tb, 0, stream>>>(zbf, WfT, bfv, X, xn, gxn, &ctrl->nrm2, 1);
    k_ctrl1<<<1, 1, 0, stream>>>(ctrl, i);
    k_dg_coef<<<gBatch, tb, 0, stream>>>(gxn, gx, xn, V, dg, lowx, ctrl);
    k_bdg<<<gBatch, tb, 0, stream>>>(dg, U, Bdg, ctrl);
    k_uv<<<gBatch, tb, 0, stream>>>(xn, x, Bdg, dg, U, V, ctrl);
    k_coef2<<<gBatch, tb, 0, stream>>>(gxn, V, ctrl);
    k_state<<<gBatch, tb, 0, stream>>>(xn, gxn, U, x, gx, upd, ctrl, (i == 0) ? 1.f : -1.f);
  }

  k_copy<<<gPlane, tb, 0, stream>>>(lowx, (float*)d_out);
}
